// MultiQueryAttention_77189152244483
// MI455X (gfx1250) — matrix operand order repaired and checked by reading, not compiled, not run on hardware
//
#include <hip/hip_runtime.h>
#include <hip/hip_bf16.h>

// Problem constants (match reference)
#define BB 2
#define SS 2048
#define DD 2048
#define HH 16
#define DHE 128   // head dim

typedef __attribute__((ext_vector_type(16))) __bf16 v16bf;
typedef __attribute__((ext_vector_type(8)))  __bf16 v8bf;
typedef __attribute__((ext_vector_type(8)))  float  v8f;

union pk2 { __bf16 h[2]; unsigned u; };

static __device__ __forceinline__ v16bf cat8(v8bf lo, v8bf hi) {
  v16bf r;
#pragma unroll
  for (int i = 0; i < 8; ++i) { r[i] = lo[i]; r[i + 8] = hi[i]; }
  return r;
}

static __device__ __forceinline__ v8bf pack8(float4 a, float4 b) {
  v8bf r;
  r[0] = (__bf16)a.x; r[1] = (__bf16)a.y; r[2] = (__bf16)a.z; r[3] = (__bf16)a.w;
  r[4] = (__bf16)b.x; r[5] = (__bf16)b.y; r[6] = (__bf16)b.z; r[7] = (__bf16)b.w;
  return r;
}

static __device__ __forceinline__ v8f wmma_bf16(v16bf a, v16bf b, v8f c) {
  // D = A(16x32 bf16) * B(32x16 bf16) + C(16x16 f32)
  return __builtin_amdgcn_wmma_f32_16x16x32_bf16(false, a, false, b, (short)0, c,
                                                 false, false);
}

// ---------------------------------------------------------------------------
// Generic GEMM: C[M,N] = A[M,K] * W[K,N] (+ bias), f32 in/out, bf16 WMMA core.
// Block tile 128x128, 256 threads = 8 wave32 (4 along M x 2 along N),
// each wave computes 32x64 = 2x4 WMMA tiles, K-step = 32.
// Double-buffered LDS with register staging: next tile's global loads are
// issued before the current tile's WMMAs, one barrier per K-step.
// ---------------------------------------------------------------------------
__global__ __launch_bounds__(256) void gemm_xw_bf16(
    const float* __restrict__ A, const float* __restrict__ W,
    const float* __restrict__ bias, float* __restrict__ C,
    int M, int N, int K) {
  __shared__ __bf16 sA[2][128][40];   // [m][k]  (pad 8 -> 16B-aligned rows)
  __shared__ __bf16 sB[2][128][40];   // [n][k]  (W transposed)

  const int tid  = threadIdx.x;
  const int lane = tid & 31;
  const int wid  = tid >> 5;       // 0..7
  const int wm   = wid >> 1;       // 0..3 -> 32-row slice
  const int wn   = wid & 1;        // 0..1 -> 64-col slice
  const int m0   = blockIdx.y * 128;
  const int n0   = blockIdx.x * 128;
  const int ln   = lane & 15;
  const int g    = lane >> 4;

  // A loader: thread -> (row, 16-col slice)
  const int arow = tid >> 1, acb = (tid & 1) * 16;
  // W loader: thread -> (k-pair, 8-col slice)  [pairs enable packed b32 stores]
  const int bk = (tid & 15) * 2;
  const int bn = (tid >> 4) * 8;

  v8f zero = {};
  v8f acc[2][4];
#pragma unroll
  for (int i = 0; i < 2; ++i)
#pragma unroll
    for (int j = 0; j < 4; ++j) acc[i][j] = zero;

  float4 ra[4];      // staged A: 16 f32
  float4 rb0[2], rb1[2];  // staged W: rows bk, bk+1, 8 f32 each

  auto gload = [&](int k0) {
    const float* ap = A + (size_t)(m0 + arow) * K + k0 + acb;
#pragma unroll
    for (int i = 0; i < 4; ++i) ra[i] = ((const float4*)ap)[i];
    const float* wp = W + (size_t)(k0 + bk) * N + n0 + bn;
    rb0[0] = ((const float4*)wp)[0];
    rb0[1] = ((const float4*)wp)[1];
    rb1[0] = ((const float4*)(wp + N))[0];
    rb1[1] = ((const float4*)(wp + N))[1];
  };

  auto stage = [&](int nb) {
    *(v8bf*)&sA[nb][arow][acb]     = pack8(ra[0], ra[1]);
    *(v8bf*)&sA[nb][arow][acb + 8] = pack8(ra[2], ra[3]);
    float r0[8] = {rb0[0].x, rb0[0].y, rb0[0].z, rb0[0].w,
                   rb0[1].x, rb0[1].y, rb0[1].z, rb0[1].w};
    float r1[8] = {rb1[0].x, rb1[0].y, rb1[0].z, rb1[0].w,
                   rb1[1].x, rb1[1].y, rb1[1].z, rb1[1].w};
#pragma unroll
    for (int c = 0; c < 8; ++c) {
      pk2 p; p.h[0] = (__bf16)r0[c]; p.h[1] = (__bf16)r1[c];
      *(unsigned*)&sB[nb][bn + c][bk] = p.u;
    }
  };

  gload(0);
  stage(0);
  __syncthreads();

  int buf = 0;
  for (int k0 = 0; k0 < K; k0 += 32) {
    const bool more = (k0 + 32) < K;
    if (more) gload(k0 + 32);   // overlap with WMMAs below

    // A-fragment: lane (m = ln, g): elems = A[m][g*16+0..15]
    v16bf af[2], bfr[4];
#pragma unroll
    for (int mt = 0; mt < 2; ++mt) {
      const __bf16* ar = &sA[buf][wm * 32 + mt * 16 + ln][0];
      af[mt] = cat8(*(const v8bf*)(ar + g * 16), *(const v8bf*)(ar + g * 16 + 8));
    }
    // B-fragment: lane (n = ln, g): elems = B[g*16+0..15][n] = sB[n][g*16+0..15]
#pragma unroll
    for (int nt = 0; nt < 4; ++nt) {
      const __bf16* br = &sB[buf][wn * 64 + nt * 16 + ln][0];
      bfr[nt] = cat8(*(const v8bf*)(br + g * 16), *(const v8bf*)(br + g * 16 + 8));
    }
#pragma unroll
    for (int mt = 0; mt < 2; ++mt)
#pragma unroll
      for (int nt = 0; nt < 4; ++nt)
        acc[mt][nt] = wmma_bf16(af[mt], bfr[nt], acc[mt][nt]);

    if (more) stage(buf ^ 1);   // other buffer: nobody reads it this step
    __syncthreads();
    buf ^= 1;
  }

  // C/D layout: vgpr r -> row (r + 8g), col = ln
#pragma unroll
  for (int mt = 0; mt < 2; ++mt) {
#pragma unroll
    for (int nt = 0; nt < 4; ++nt) {
      const int col = n0 + wn * 64 + nt * 16 + ln;
      const float bv = bias ? bias[col] : 0.0f;
#pragma unroll
      for (int r = 0; r < 8; ++r) {
        const int row = m0 + wm * 32 + mt * 16 + g * 8 + r;
        C[(size_t)row * N + col] = acc[mt][nt][r] + bv;
      }
    }
  }
}

// ---------------------------------------------------------------------------
// Flash attention (causal, MQA: 1 KV head shared by 16 Q heads).
// Grid: (S/64, H, B); 128 threads = 4 wave32, each wave owns 16 query rows.
// Q fragments live in registers; K tile [64x128] and V^T [128x64] staged in
// LDS as bf16 (packed b128/b32 stores); online softmax with shfl_xor row
// reductions; P re-laid-out through a per-wave LDS patch for the P*V WMMAs.
// ---------------------------------------------------------------------------
__global__ __launch_bounds__(128) void mqa_flash_fwd(
    const float* __restrict__ Q,   // [B,S,H,DH]
    const float* __restrict__ Kp,  // [B,S,DH]
    const float* __restrict__ Vp,  // [B,S,DH]
    float* __restrict__ Oa)        // [B,S,H,DH]
{
  __shared__ __bf16 sK[64][136];       // [key][dh]
  __shared__ __bf16 sVt[128][72];      // [dh][key]
  __shared__ __bf16 sP[4][16][72];     // per-wave P patch [qrow][key]

  const int qt = blockIdx.x, h = blockIdx.y, b = blockIdx.z;
  const int tid = threadIdx.x;
  const int lane = tid & 31, wid = tid >> 5;
  const int ln = lane & 15, g = lane >> 4;
  const int q0 = qt * 64;
  const float qscale = 0.08838834764831845f;  // 1/sqrt(128)

  // K loader: thread -> (key row, 64-dh half)
  const int krow = tid >> 1, kcb = (tid & 1) * 64;
  // V loader: thread -> (key pair, 32-dh slice)  [pairs -> packed b32 stores]
  const int vkp = (tid & 31) * 2;
  const int vdb = (tid >> 5) * 32;

  // Load Q as 4 A-fragments (K-chunks of 32 over DH=128), scaled, bf16.
  v16bf qa[4];
  {
    const float* qp = Q + (((size_t)b * SS + q0 + wid * 16 + ln) * HH + h) * DHE;
#pragma unroll
    for (int c = 0; c < 4; ++c) {
      float t[16];
      const float4 f0 = *(const float4*)(qp + c * 32 + g * 8);
      const float4 f1 = *(const float4*)(qp + c * 32 + g * 8 + 4);
      const float4 f2 = *(const float4*)(qp + c * 32 + 16 + g * 8);
      const float4 f3 = *(const float4*)(qp + c * 32 + 16 + g * 8 + 4);
      t[0]=f0.x; t[1]=f0.y; t[2]=f0.z; t[3]=f0.w;
      t[4]=f1.x; t[5]=f1.y; t[6]=f1.z; t[7]=f1.w;
      t[8]=f2.x; t[9]=f2.y; t[10]=f2.z; t[11]=f2.w;
      t[12]=f3.x; t[13]=f3.y; t[14]=f3.z; t[15]=f3.w;
      v16bf a;
#pragma unroll
      for (int i = 0; i < 16; ++i) a[i] = (__bf16)(t[i] * qscale);
      qa[c] = a;
    }
  }

  v8f zero = {};
  v8f oacc[8];
#pragma unroll
  for (int t = 0; t < 8; ++t) oacc[t] = zero;
  float mrow[8], lrow[8];
#pragma unroll
  for (int r = 0; r < 8; ++r) { mrow[r] = -3.0e38f; lrow[r] = 0.0f; }

  for (int kt = 0; kt <= qt; ++kt) {
    const int k0 = kt * 64;
    {
      // K tile: vector b128 LDS stores (8 bf16 each)
      const float* kpr = Kp + ((size_t)b * SS + k0 + krow) * DHE + kcb;
#pragma unroll
      for (int i = 0; i < 8; ++i) {
        float4 fa = ((const float4*)kpr)[2 * i];
        float4 fb = ((const float4*)kpr)[2 * i + 1];
        *(v8bf*)&sK[krow][kcb + 8 * i] = pack8(fa, fb);
      }
      // V^T tile: key pairs -> packed b32 stores along the key axis
      const float* vp0 = Vp + ((size_t)b * SS + k0 + vkp) * DHE + vdb;
      const float* vp1 = vp0 + DHE;
      float f0[32], f1[32];
#pragma unroll
      for (int i = 0; i < 8; ++i) {
        float4 a0 = ((const float4*)vp0)[i];
        float4 a1 = ((const float4*)vp1)[i];
        f0[4*i+0]=a0.x; f0[4*i+1]=a0.y; f0[4*i+2]=a0.z; f0[4*i+3]=a0.w;
        f1[4*i+0]=a1.x; f1[4*i+1]=a1.y; f1[4*i+2]=a1.z; f1[4*i+3]=a1.w;
      }
#pragma unroll
      for (int c = 0; c < 32; ++c) {
        pk2 p; p.h[0] = (__bf16)f0[c]; p.h[1] = (__bf16)f1[c];
        *(unsigned*)&sVt[vdb + c][vkp] = p.u;
      }
    }
    __syncthreads();

    // S = Q * K^T : B-frag elements = sK[key=n][c*32 + g*8 + 0..7], sK[key=n][c*32 + 16 + g*8 + 0..7]
    v8f sc[4];
#pragma unroll
    for (int nt = 0; nt < 4; ++nt) {
      v8f s = zero;
#pragma unroll
      for (int c = 0; c < 4; ++c) {
        const __bf16* br = &sK[nt * 16 + ln][c * 32 + g * 8];
        s = wmma_bf16(qa[c], cat8(*(const v8bf*)br, *(const v8bf*)(br + 16)), s);
      }
      sc[nt] = s;
    }

    // Causal mask + row max (rows live in 16-lane halves; xor masks 1..8)
    float rmax[8];
#pragma unroll
    for (int r = 0; r < 8; ++r) {
      const int qi = q0 + wid * 16 + g * 8 + r;
      float mx = -3.0e38f;
#pragma unroll
      for (int nt = 0; nt < 4; ++nt) {
        const int ki = k0 + nt * 16 + ln;
        float v = sc[nt][r];
        v = (ki > qi) ? -3.0e38f : v;
        sc[nt][r] = v;
        mx = fmaxf(mx, v);
      }
      rmax[r] = mx;
    }
#pragma unroll
    for (int r = 0; r < 8; ++r)
#pragma unroll
      for (int d = 1; d < 16; d <<= 1)
        rmax[r] = fmaxf(rmax[r], __shfl_xor(rmax[r], d, 32));

#pragma unroll
    for (int r = 0; r < 8; ++r) {
      const float mnew = fmaxf(mrow[r], rmax[r]);
      const float sf = __expf(mrow[r] - mnew);
      mrow[r] = mnew;
      float rs = 0.0f;
#pragma unroll
      for (int nt = 0; nt < 4; ++nt) {
        const float p = __expf(sc[nt][r] - mnew);
        sc[nt][r] = p;
        rs += p;
      }
#pragma unroll
      for (int d = 1; d < 16; d <<= 1) rs += __shfl_xor(rs, d, 32);
      lrow[r] = lrow[r] * sf + rs;
#pragma unroll
      for (int t = 0; t < 8; ++t) oacc[t][r] *= sf;
    }

    // P (accumulator layout) -> per-wave LDS patch -> A-fragment layout
#pragma unroll
    for (int nt = 0; nt < 4; ++nt)
#pragma unroll
      for (int r = 0; r < 8; ++r)
        sP[wid][g * 8 + r][nt * 16 + ln] = (__bf16)sc[nt][r];

    // O += P * V : A = P (16x64, two K=32 chunks), B from sVt (contiguous)
#pragma unroll
    for (int c2 = 0; c2 < 2; ++c2) {
      const __bf16* pr = &sP[wid][ln][c2 * 32];
      const v16bf pa =
          cat8(*(const v8bf*)(pr + g * 16), *(const v8bf*)(pr + g * 16 + 8));
#pragma unroll
      for (int t = 0; t < 8; ++t) {
        const __bf16* vr = &sVt[t * 16 + ln][c2 * 32 + g * 16];
        oacc[t] =
            wmma_bf16(pa, cat8(*(const v8bf*)vr, *(const v8bf*)(vr + 8)), oacc[t]);
      }
    }
    __syncthreads();
  }

  // Normalize and store attn as [b, s, h, dh] so the O-proj is a plain GEMM.
  float inv[8];
#pragma unroll
  for (int r = 0; r < 8; ++r) inv[r] = 1.0f / lrow[r];
#pragma unroll
  for (int t = 0; t < 8; ++t)
#pragma unroll
    for (int r = 0; r < 8; ++r) {
      const int qi = q0 + wid * 16 + g * 8 + r;
      Oa[(((size_t)b * SS + qi) * HH + h) * DHE + t * 16 + ln] =
          oacc[t][r] * inv[r];
    }
}

// ---------------------------------------------------------------------------
extern "C" void kernel_launch(void* const* d_in, const int* in_sizes, int n_in,
                              void* d_out, int out_size, void* d_ws, size_t ws_size,
                              hipStream_t stream) {
  (void)in_sizes; (void)n_in; (void)out_size; (void)ws_size;
  const float* x  = (const float*)d_in[0];
  // d_in[1] = boolean mask: unused, causal structure computed analytically
  const float* Wq = (const float*)d_in[2];   // [D, H*DH] flat
  const float* Wk = (const float*)d_in[3];   // [D, DH]
  const float* Wv = (const float*)d_in[4];   // [D, DH]
  const float* Wo = (const float*)d_in[5];   // [H*DH, D] flat
  const float* bo = (const float*)d_in[6];   // [D]
  float* out = (float*)d_out;

  float* q_ws = (float*)d_ws;                          // B*S*H*DH f32 (32 MB)
  float* k_ws = q_ws + (size_t)BB * SS * HH * DHE;     // B*S*DH     (2 MB)
  float* v_ws = k_ws + (size_t)BB * SS * DHE;          // B*S*DH     (2 MB)
  float* a_ws = v_ws + (size_t)BB * SS * DHE;          // B*S*H*DH   (32 MB)

  const int M = BB * SS;  // 4096
  dim3 blk(256);

  // Projections
  gemm_xw_bf16<<<dim3(DD / 128, M / 128), blk, 0, stream>>>(x, Wq, nullptr, q_ws,
                                                            M, HH * DHE, DD);
  gemm_xw_bf16<<<dim3(DHE / 128, M / 128), blk, 0, stream>>>(x, Wk, nullptr, k_ws,
                                                             M, DHE, DD);
  gemm_xw_bf16<<<dim3(DHE / 128, M / 128), blk, 0, stream>>>(x, Wv, nullptr, v_ws,
                                                             M, DHE, DD);
  // Attention
  mqa_flash_fwd<<<dim3(SS / 64, HH, BB), dim3(128), 0, stream>>>(q_ws, k_ws, v_ws,
                                                                 a_ws);
  // Output projection with bias
  gemm_xw_bf16<<<dim3(DD / 128, M / 128), blk, 0, stream>>>(a_ws, Wo, bo, out, M,
                                                            DD, HH * DHE);
}
